// LocPosPPM4_35201551958663
// MI455X (gfx1250) — matrix operand order repaired and checked by reading, not compiled, not run on hardware
//
#include <hip/hip_runtime.h>
#include <hip/hip_bf16.h>

// ---------------- problem constants ----------------
#define NB   8
#define CH   256
#define IMW  56
#define HW   3136      // 56*56
#define NP   512
#define LSZ  7
#define LS2  49

typedef __bf16 bf16_t;
typedef __attribute__((ext_vector_type(16))) __bf16 v16bf;
typedef __attribute__((ext_vector_type(8)))  __bf16 v8bf;
typedef __attribute__((ext_vector_type(8)))  float  v8f;
typedef __attribute__((ext_vector_type(4)))  unsigned int u32x4;
typedef __attribute__((ext_vector_type(8)))  int i32x8;
typedef __attribute__((ext_vector_type(4)))  int i32x4;

// =====================================================================
// 0) float -> bf16 weight conversion
// =====================================================================
__global__ void cvt_bf16_kernel(const float* __restrict__ src,
                                bf16_t* __restrict__ dst, int n) {
    int i = blockIdx.x * blockDim.x + threadIdx.x;
    if (i < n) dst[i] = (bf16_t)src[i];
}

// =====================================================================
// 1) x[n][c][pix] -> xT_bf[n][pix][c] (bf16) + per-pixel inverse L2 norm
// =====================================================================
__global__ void prep_x_kernel(const float* __restrict__ x,
                              bf16_t* __restrict__ xT,
                              float* __restrict__ ixn) {
    __shared__ float tile[32][257];
    __shared__ float part[8][32];
    int n    = blockIdx.y;
    int pix0 = blockIdx.x * 32;
    int wv = threadIdx.x >> 5;
    int l  = threadIdx.x & 31;
    const float* xb = x + (size_t)n * CH * HW;

    for (int c = wv; c < CH; c += 8)
        tile[l][c] = xb[(size_t)c * HW + pix0 + l];
    __syncthreads();

    float s = 0.f;
    #pragma unroll 8
    for (int j = 0; j < 32; ++j) { float v = tile[l][wv * 32 + j]; s += v * v; }
    part[wv][l] = s;
    __syncthreads();
    if (wv == 0) {
        float t = 0.f;
        #pragma unroll
        for (int j = 0; j < 8; ++j) t += part[j][l];
        float nr = sqrtf(t);
        ixn[(size_t)n * HW + pix0 + l] = 1.0f / fmaxf(nr, 1e-12f);
    }
    __syncthreads();

    for (int pl = 0; pl < 32; ++pl)
        xT[((size_t)n * HW + pix0 + pl) * CH + threadIdx.x] = (bf16_t)tile[pl][threadIdx.x];
}

// =====================================================================
// TDM: issue a 2-D tensor_load_to_lds of a [64 rows x 256 bf16] tile.
// Descriptor per CDNA5 ISA ch.8: data_size=1 (2 bytes), tensor_dim0=256,
// tile_dim0=256, tile_dim1=64, tensor_dim0_stride=256, type=2 ("image").
// This toolchain exposes the 6-arg builtin (groups 0..3 + extra group + cpol).
// =====================================================================
__device__ __forceinline__ void tdm_load_panel_64x256(unsigned int lds_off,
                                                      const void* gptr,
                                                      unsigned int rows_total) {
    unsigned long long ga = (unsigned long long)(uintptr_t)gptr;
    u32x4 g0;
    g0[0] = 1u;                                   // count=1, user mode
    g0[1] = lds_off;                              // lds_addr (bytes)
    g0[2] = (unsigned int)ga;                     // global_addr[31:0]
    g0[3] = (unsigned int)((ga >> 32) & 0x01FFFFFFu) | 0x80000000u; // [56:32] | type=2

    i32x8 g1;
    g1[0] = (int)0x00010000;                      // workgroup_mask=0, data_size=1 (2B)
    g1[1] = (int)(256u << 16);                    // atomic_barrier=0, tensor_dim0.lo16=256
    g1[2] = (int)((rows_total & 0xFFFFu) << 16);  // tensor_dim0.hi16=0, tensor_dim1.lo16
    g1[3] = (int)((rows_total >> 16) | (256u << 16)); // tensor_dim1.hi16, tile_dim0=256
    g1[4] = 64;                                   // tile_dim1=64, tile_dim2=0
    g1[5] = 256;                                  // tensor_dim0_stride.lo32 = 256
    g1[6] = 0;                                    // stride.hi16, dim1_stride.lo16
    g1[7] = 0;

    i32x4 gz4 = {0, 0, 0, 0};
    i32x8 gz8 = {0, 0, 0, 0, 0, 0, 0, 0};
    __builtin_amdgcn_tensor_load_to_lds(g0, g1, gz4, gz4, gz8, 0);
}

// =====================================================================
// WMMA fragment loaders from LDS panels (ISA 7.12.2 layouts, wave32)
// =====================================================================
__device__ __forceinline__ v16bf lds_a_frag(const bf16_t* tA, int row0, int k0, int lane) {
    int row = row0 + (lane & 15);
    int g   = lane >> 4;
    const bf16_t* p0 = tA + row * 256 + k0 + g * 8;   // j=0..7  : k = g*8+j
    v8bf lo = *(const v8bf*)p0;                       // j=8..15 : k = 16+g*8+j
    v8bf hi = *(const v8bf*)(p0 + 16);
    v16bf r;
    #pragma unroll
    for (int j = 0; j < 8; ++j) { r[j] = lo[j]; r[j + 8] = hi[j]; }
    return r;
}

__device__ __forceinline__ v16bf lds_b_frag(const bf16_t* tB, int col0, int k0, int lane) {
    int col = col0 + (lane & 15);
    int g   = lane >> 4;                              // j=0..7: k = g*8+j ; j=8..15: k = 16+g*8+j
    const bf16_t* p0 = tB + col * 256 + k0 + g * 8;
    v8bf lo = *(const v8bf*)p0;
    v8bf hi = *(const v8bf*)(p0 + 16);
    v16bf r;
    #pragma unroll
    for (int j = 0; j < 8; ++j) { r[j] = lo[j]; r[j + 8] = hi[j]; }
    return r;
}

// =====================================================================
// 2/5) GEMM: out = W[256x256] * BT^T per image, K=256 staged via TDM.
//  Block computes 64 o x 64 pix. Wave = 2 o-tiles x 1 pix-tile (B reuse).
//  MODE 0: out[n][o][pix]  ;  MODE 1: out[n][pix][o] (v8f stores)
//  grid = (CH/64, HW/64, NB), block = 256 (8 waves)
// =====================================================================
template <int MODE>
__global__ void gemm_wmma_tdm_kernel(const bf16_t* __restrict__ W,
                                     const bf16_t* __restrict__ BT,
                                     const float* __restrict__ bias,
                                     float* __restrict__ out) {
    __shared__ __align__(128) bf16_t tileA[64][256];   // 32 KB weight panel
    __shared__ __align__(128) bf16_t tileB[64][256];   // 32 KB pixel panel

    int n      = blockIdx.z;
    int o_base = blockIdx.x * 64;
    int p_base = blockIdx.y * 64;
    int wave = threadIdx.x >> 5;
    int lane = threadIdx.x & 31;

    if (wave == 0) {
        unsigned int ldsA = (unsigned int)(uintptr_t)&tileA[0][0];
        unsigned int ldsB = (unsigned int)(uintptr_t)&tileB[0][0];
        tdm_load_panel_64x256(ldsA, W + (size_t)o_base * CH, CH);
        tdm_load_panel_64x256(ldsB, BT + ((size_t)n * HW + p_base) * CH,
                              (unsigned int)(NB * HW));
        __builtin_amdgcn_s_wait_tensorcnt(0);
    }
    __syncthreads();

    int pt = wave & 3;            // pixel tile 0..3
    int op = wave >> 2;           // o-tile pair 0..1
    int pix0_l = pt * 16;
    int o0_l   = op * 32;

    v8f acc0 = {0.f, 0.f, 0.f, 0.f, 0.f, 0.f, 0.f, 0.f};
    v8f acc1 = {0.f, 0.f, 0.f, 0.f, 0.f, 0.f, 0.f, 0.f};
    #pragma unroll
    for (int kb = 0; kb < 8; ++kb) {
        int k0 = kb * 32;
        v16bf b  = lds_b_frag(&tileB[0][0], pix0_l, k0, lane);
        v16bf a0 = lds_a_frag(&tileA[0][0], o0_l,      k0, lane);
        v16bf a1 = lds_a_frag(&tileA[0][0], o0_l + 16, k0, lane);
        acc0 = __builtin_amdgcn_wmma_f32_16x16x32_bf16(false, a0, false, b,
                                                       (short)0, acc0, false, false);
        acc1 = __builtin_amdgcn_wmma_f32_16x16x32_bf16(false, a1, false, b,
                                                       (short)0, acc1, false, false);
    }

    int g   = lane >> 4;
    int col = lane & 15;
    int pix = p_base + pix0_l + col;

    #pragma unroll
    for (int t = 0; t < 2; ++t) {
        const v8f& acc = t ? acc1 : acc0;
        int o0g = o_base + o0_l + t * 16;
        if (MODE == 0) {
            #pragma unroll
            for (int r = 0; r < 8; ++r) {
                int o = o0g + 8 * g + r;              // D row M = r + 8*g
                out[((size_t)n * CH + o) * HW + pix] = acc[r] + bias[o];
            }
        } else {
            v8f v;
            #pragma unroll
            for (int r = 0; r < 8; ++r) v[r] = acc[r] + bias[o0g + 8 * g + r];
            float* dst = out + ((size_t)n * HW + pix) * CH + o0g + 8 * g;
            *(v8f*)dst = v;                            // 32B contiguous per lane
        }
    }
}

// =====================================================================
// 3) BN training-mode batch stats per channel -> fused scale/shift
// =====================================================================
__global__ void bn_stats_kernel(const float* __restrict__ y,
                                const float* __restrict__ bn_g,
                                const float* __restrict__ bn_b,
                                float* __restrict__ scale,
                                float* __restrict__ shift) {
    int c = blockIdx.x;
    int tid = threadIdx.x;
    float s = 0.f, s2 = 0.f;
    for (int n = 0; n < NB; ++n) {
        const float* p = y + ((size_t)n * CH + c) * HW;
        for (int i = tid; i < HW; i += 256) { float v = p[i]; s += v; s2 += v * v; }
    }
    __shared__ float sh1[256], sh2[256];
    sh1[tid] = s; sh2[tid] = s2;
    __syncthreads();
    for (int off = 128; off > 0; off >>= 1) {
        if (tid < off) { sh1[tid] += sh1[tid + off]; sh2[tid] += sh2[tid + off]; }
        __syncthreads();
    }
    if (tid == 0) {
        const float cnt = (float)NB * (float)HW;
        float mu  = sh1[0] / cnt;
        float var = sh2[0] / cnt - mu * mu;
        float rs  = rsqrtf(var + 1e-5f);
        float sc  = bn_g[c] * rs;
        scale[c] = sc;
        shift[c] = bn_b[c] - mu * sc;
    }
}

// =====================================================================
// 4) BN-apply + ReLU + transpose to bf16  [n][c][pix] -> [n][pix][c]
// =====================================================================
__global__ void act_transpose_kernel(const float* __restrict__ y,
                                     const float* __restrict__ scale,
                                     const float* __restrict__ shift,
                                     bf16_t* __restrict__ yT) {
    __shared__ float tile[32][257];
    int n    = blockIdx.y;
    int pix0 = blockIdx.x * 32;
    int wv = threadIdx.x >> 5;
    int l  = threadIdx.x & 31;
    const float* yb = y + (size_t)n * CH * HW;
    for (int c = wv; c < CH; c += 8) {
        float v = yb[(size_t)c * HW + pix0 + l];
        v = fmaxf(v * scale[c] + shift[c], 0.f);
        tile[l][c] = v;
    }
    __syncthreads();
    for (int pl = 0; pl < 32; ++pl)
        yT[((size_t)n * HW + pix0 + pl) * CH + threadIdx.x] = (bf16_t)tile[pl][threadIdx.x];
}

// =====================================================================
// 6) per-pixel inverse L2 norm of tT[n][pix][c]; one wave per pixel
// =====================================================================
__global__ void inv_norm_t_kernel(const float* __restrict__ tT,
                                  float* __restrict__ itn) {
    int n = blockIdx.y;
    int pix = blockIdx.x * 8 + (threadIdx.x >> 5);
    int lane = threadIdx.x & 31;
    const float* p = tT + ((size_t)n * HW + pix) * CH + lane * 8;
    v8f v = *(const v8f*)p;
    float s = 0.f;
    #pragma unroll
    for (int j = 0; j < 8; ++j) s += v[j] * v[j];
    #pragma unroll
    for (int m = 16; m >= 1; m >>= 1) s += __shfl_xor(s, m, 32);
    if (lane == 0)
        itn[(size_t)n * HW + pix] = 1.0f / fmaxf(sqrtf(s), 1e-12f);
}

// =====================================================================
// 7) gather + power-law similarity + weighted accumulation
// =====================================================================
__global__ void final_gather_kernel(const bf16_t* __restrict__ xT,
                                    const float* __restrict__ ixn,
                                    const float* __restrict__ tT,
                                    const float* __restrict__ itn,
                                    const int* __restrict__ pos_idx,
                                    float* __restrict__ out) {
    __shared__ float qs[256];
    __shared__ float simw[LS2];
    __shared__ int   wp[LS2];
    int n = blockIdx.y;
    int p = blockIdx.x;
    int tid = threadIdx.x;

    const int* pi = pos_idx + ((size_t)n * NP + p) * 2;
    int qy = pi[0], qx = pi[1];
    int qpix = qy * IMW + qx;
    int sy = (qy / LSZ) * LSZ, sx = (qx / LSZ) * LSZ;

    const bf16_t* xb = xT + (size_t)n * HW * CH;
    float invq = ixn[(size_t)n * HW + qpix];
    qs[tid] = (float)xb[(size_t)qpix * CH + tid] * invq;
    __syncthreads();

    int wave = tid >> 5, lane = tid & 31;
    for (int k = wave; k < LS2; k += 8) {
        int oy = k % LSZ, ox = k / LSZ;          // x-major window order
        int wpix = (sy + oy) * IMW + (sx + ox);
        const bf16_t* xw = xb + (size_t)wpix * CH;
        float d = 0.f;
        #pragma unroll
        for (int j = 0; j < 8; ++j)
            d += qs[lane * 8 + j] * (float)xw[lane * 8 + j];
        #pragma unroll
        for (int m = 16; m >= 1; m >>= 1) d += __shfl_xor(d, m, 32);
        if (lane == 0) {
            d *= ixn[(size_t)n * HW + wpix];
            d = fmaxf(d, 0.f);
            simw[k] = d * d * itn[(size_t)n * HW + wpix];  // fold tx norm in
            wp[k] = wpix;
        }
    }
    __syncthreads();

    float acc = 0.f;
    const float* tb = tT + (size_t)n * HW * CH;
    #pragma unroll 7
    for (int k = 0; k < LS2; ++k)
        acc += simw[k] * tb[(size_t)wp[k] * CH + tid];

    out[((size_t)n * CH + tid) * NP + p] = acc;   // [n, c, p]
}

// =====================================================================
// host launcher
// =====================================================================
static inline size_t align256(size_t v) { return (v + 255) & ~(size_t)255; }

extern "C" void kernel_launch(void* const* d_in, const int* in_sizes, int n_in,
                              void* d_out, int out_size, void* d_ws, size_t ws_size,
                              hipStream_t stream) {
    const float* x       = (const float*)d_in[0];
    const int*   pos_idx = (const int*)  d_in[1];
    const float* w1      = (const float*)d_in[2];
    const float* b1      = (const float*)d_in[3];
    const float* bn_g    = (const float*)d_in[4];
    const float* bn_b    = (const float*)d_in[5];
    const float* w2      = (const float*)d_in[6];
    const float* b2      = (const float*)d_in[7];
    float* out = (float*)d_out;

    // workspace carve-up
    char* ws = (char*)d_ws;
    size_t off = 0;
    bf16_t* w1_bf = (bf16_t*)(ws + off); off = align256(off + (size_t)CH * CH * sizeof(bf16_t));
    bf16_t* w2_bf = (bf16_t*)(ws + off); off = align256(off + (size_t)CH * CH * sizeof(bf16_t));
    bf16_t* xT_bf = (bf16_t*)(ws + off); off = align256(off + (size_t)NB * HW * CH * sizeof(bf16_t));
    float*  ixn   = (float*) (ws + off); off = align256(off + (size_t)NB * HW * sizeof(float));
    float*  ybuf  = (float*) (ws + off); off = align256(off + (size_t)NB * CH * HW * sizeof(float));
    bf16_t* yT_bf = (bf16_t*)(ws + off); off = align256(off + (size_t)NB * HW * CH * sizeof(bf16_t));
    float*  scale = (float*) (ws + off); off = align256(off + (size_t)CH * sizeof(float));
    float*  shift = (float*) (ws + off); off = align256(off + (size_t)CH * sizeof(float));
    float*  tT    = (float*) (ws + off); off = align256(off + (size_t)NB * HW * CH * sizeof(float));
    float*  itn   = (float*) (ws + off); off = align256(off + (size_t)NB * HW * sizeof(float));
    (void)ws_size; (void)in_sizes; (void)n_in; (void)out_size;

    // 0) weights -> bf16
    cvt_bf16_kernel<<<(CH * CH + 255) / 256, 256, 0, stream>>>(w1, w1_bf, CH * CH);
    cvt_bf16_kernel<<<(CH * CH + 255) / 256, 256, 0, stream>>>(w2, w2_bf, CH * CH);

    // 1) transpose x + inverse norms
    prep_x_kernel<<<dim3(HW / 32, NB), 256, 0, stream>>>(x, xT_bf, ixn);

    // 2) GEMM1: y = w1 * x + b1   -> [n][c][pix]   (TDM-staged panels)
    gemm_wmma_tdm_kernel<0><<<dim3(CH / 64, HW / 64, NB), 256, 0, stream>>>(w1_bf, xT_bf, b1, ybuf);

    // 3) BN batch stats -> fused scale/shift
    bn_stats_kernel<<<CH, 256, 0, stream>>>(ybuf, bn_g, bn_b, scale, shift);

    // 4) BN + ReLU + transpose -> bf16 [n][pix][c]
    act_transpose_kernel<<<dim3(HW / 32, NB), 256, 0, stream>>>(ybuf, scale, shift, yT_bf);

    // 5) GEMM2: t = w2 * y_act + b2 -> transposed [n][pix][c]
    gemm_wmma_tdm_kernel<1><<<dim3(CH / 64, HW / 64, NB), 256, 0, stream>>>(w2_bf, yT_bf, b2, tT);

    // 6) inverse norms of t
    inv_norm_t_kernel<<<dim3(HW / 8, NB), 256, 0, stream>>>(tT, itn);

    // 7) gather + similarity + accumulation
    final_gather_kernel<<<dim3(NP, NB), 256, 0, stream>>>(xT_bf, ixn, tT, itn, pos_idx, out);
}
